// FluxRoPEAttention_47287589929115
// MI455X (gfx1250) — matrix operand order repaired and checked by reading, not compiled, not run on hardware
//
#include <hip/hip_runtime.h>
#include <math.h>

#define B_   2
#define H_   24
#define L_   2048
#define D_   128
#define KT   32            // key positions per tile
#define NKT  (L_ / KT)     // 64
#define QB   128           // query rows per workgroup (4 waves x 32)
#define NQB  (L_ / QB)     // 16

typedef _Float16 h16 __attribute__((ext_vector_type(16)));
typedef _Float16 h8  __attribute__((ext_vector_type(8)));
typedef float    f8  __attribute__((ext_vector_type(8)));
typedef float    f4  __attribute__((ext_vector_type(4)));

// XOR-butterfly max across the 16-lane C-layout row groups using
// v_permlane16_b32 (VALU cross-lane, keeps the DS pipe free for tile loads).
static __device__ __forceinline__ float pl16(float v, unsigned s0, unsigned s1) {
  unsigned iv = __builtin_bit_cast(unsigned, v);
  unsigned r  = __builtin_amdgcn_permlane16(iv, iv, s0, s1, false, false);
  return __builtin_bit_cast(float, r);
}
static __device__ __forceinline__ float red_max16(float v) {
  v = fmaxf(v, pl16(v, 0x67452301u, 0xEFCDAB89u));  // xor 1
  v = fmaxf(v, pl16(v, 0x54761032u, 0xDCFE98BAu));  // xor 2
  v = fmaxf(v, pl16(v, 0x32107654u, 0xBA98FEDCu));  // xor 4
  v = fmaxf(v, pl16(v, 0xFEDCBA98u, 0x76543210u));  // xor 8
  return v;
}

__global__ __launch_bounds__(128) void flux_rope_attn_kernel(
    const float* __restrict__ q, const float* __restrict__ k,
    const float* __restrict__ v, const float* __restrict__ pe,
    float* __restrict__ out) {
  __shared__ __align__(32) _Float16 sK[KT * D_];        // rope'd K, [kpos][d]
  __shared__ __align__(32) _Float16 sVt[D_ * KT];       // V transposed, [d][kpos]
  __shared__ __align__(32) _Float16 sP[4][2][16 * 32];  // per-wave/slab P scratch

  const int tid  = threadIdx.x;
  const int wid  = tid >> 5;
  const int lane = tid & 31;
  const int hi   = lane >> 4;   // half-wave select
  const int ml   = lane & 15;   // M (A) or N (B/C) coordinate

  const int bh   = blockIdx.x / NQB;   // b*H + h
  const int qblk = blockIdx.x % NQB;
  const int b    = bh / H_;
  const int h    = bh % H_;
  const int q0   = qblk * QB + wid * 32;   // 32 query rows per wave (2 slabs)

  // fold 1/sqrt(128) AND log2(e) into Q so softmax runs in exp2 domain:
  // 2^(s' - m') with s' = s*log2(e) equals e^(s - m) exactly.
  const float scale = 0.08838834764831845f * 1.4426950408889634f;

  // all-ones B matrix for row-sum WMMA (sum over N of P lands in C[r] per lane)
  h16 ones;
#pragma unroll
  for (int i = 0; i < 16; ++i) ones[i] = (_Float16)1.0f;

  // ---- load 2x16-row Q slabs with RoPE, directly into WMMA A layout ----
  // f16 A 16x32: lanes 0-15 (M=lane): elems 0..7 -> K 0..7, 8..15 -> K 16..23
  //              lanes 16-31 (M=lane-16): elems 0..7 -> K 8..15, 8..15 -> K 24..31
  h16 qa[2][4];
#pragma unroll
  for (int s = 0; s < 2; ++s) {
    const int    qrow = q0 + s * 16 + ml;
    const float* qp   = q  + ((size_t)bh * L_ + qrow) * D_;
    const float* pr   = pe + (size_t)qrow * (D_ * 2);   // (D/2)*4 floats per row
#pragma unroll
    for (int dc = 0; dc < 4; ++dc) {
#pragma unroll
      for (int seg = 0; seg < 2; ++seg) {
        const int d0 = dc * 32 + seg * 16 + hi * 8;
        f4 x0 = *(const f4*)(qp + d0);
        f4 x1 = *(const f4*)(qp + d0 + 4);
        float xs[8] = {x0[0], x0[1], x0[2], x0[3], x1[0], x1[1], x1[2], x1[3]};
#pragma unroll
        for (int p = 0; p < 4; ++p) {
          f4 c = *(const f4*)(pr + (size_t)(d0 / 2 + p) * 4);
          float a0 = xs[2 * p], a1 = xs[2 * p + 1];
          qa[s][dc][seg * 8 + 2 * p]     = (_Float16)(scale * (c[0] * a0 + c[1] * a1));
          qa[s][dc][seg * 8 + 2 * p + 1] = (_Float16)(scale * (c[2] * a0 + c[3] * a1));
        }
      }
    }
  }

  f8 o[2][8] = {};              // O accumulators: 2 slabs x 8 d-tiles
  float mrow[2][8], lrow[2][8];
#pragma unroll
  for (int s = 0; s < 2; ++s)
#pragma unroll
    for (int r = 0; r < 8; ++r) { mrow[s][r] = -3.0e38f; lrow[s][r] = 0.0f; }

  for (int kt = 0; kt < NKT; ++kt) {
    // ---- cooperative staging: RoPE'd K (row-major f16) ----
    {
      const int    kloc = tid >> 2;
      const int    d0   = (tid & 3) * 32;
      const int    krow = kt * KT + kloc;
      const float* kp   = k  + ((size_t)bh * L_ + krow) * D_ + d0;
      const float* pr   = pe + (size_t)krow * (D_ * 2) + (size_t)d0 * 2;
      _Float16 tmp[32];
#pragma unroll
      for (int p = 0; p < 16; ++p) {
        float a0 = kp[2 * p], a1 = kp[2 * p + 1];
        f4 c = *(const f4*)(pr + p * 4);
        tmp[2 * p]     = (_Float16)(c[0] * a0 + c[1] * a1);
        tmp[2 * p + 1] = (_Float16)(c[2] * a0 + c[3] * a1);
      }
#pragma unroll
      for (int i = 0; i < 4; ++i)
        *(h8*)(&sK[kloc * D_ + d0 + i * 8]) = *(const h8*)(&tmp[i * 8]);
      if (kt + 1 < NKT) __builtin_prefetch(kp + KT * D_, 0, 1);
    }
    // ---- cooperative staging: V transposed (f16) ----
    {
      const int    d  = tid;
      const float* vp = v + ((size_t)bh * L_ + (size_t)kt * KT) * D_ + d;
      _Float16 tmp[KT];
#pragma unroll
      for (int j = 0; j < KT; ++j) tmp[j] = (_Float16)vp[(size_t)j * D_];
#pragma unroll
      for (int i = 0; i < 4; ++i)
        *(h8*)(&sVt[d * KT + i * 8]) = *(const h8*)(&tmp[i * 8]);
      if (kt + 1 < NKT) __builtin_prefetch(vp + KT * D_, 0, 1);
    }
    __syncthreads();

    // ---- S = Q . K^T : 8 shared B tiles feed both A slabs (16 WMMAs) ----
    h16 bt[8];
#pragma unroll
    for (int dc = 0; dc < 4; ++dc) {
      h8 b0l = *(const h8*)(&sK[ml * D_        + dc * 32 + hi * 8]);
      h8 b0h = *(const h8*)(&sK[ml * D_        + dc * 32 + 16 + hi * 8]);
      h8 b1l = *(const h8*)(&sK[(16 + ml) * D_ + dc * 32 + hi * 8]);
      h8 b1h = *(const h8*)(&sK[(16 + ml) * D_ + dc * 32 + 16 + hi * 8]);
#pragma unroll
      for (int i = 0; i < 8; ++i) {
        bt[dc][i] = b0l[i]; bt[dc][8 + i] = b0h[i];
        bt[4 + dc][i] = b1l[i]; bt[4 + dc][8 + i] = b1h[i];
      }
    }
    f8 c0[2] = {}, c1[2] = {};
#pragma unroll
    for (int dc = 0; dc < 4; ++dc) {
#pragma unroll
      for (int s = 0; s < 2; ++s) {
        c0[s] = __builtin_amdgcn_wmma_f32_16x16x32_f16(false, qa[s][dc], false, bt[dc],
                                                       (short)0, c0[s], false, false);
        c1[s] = __builtin_amdgcn_wmma_f32_16x16x32_f16(false, qa[s][dc], false, bt[4 + dc],
                                                       (short)0, c1[s], false, false);
      }
    }

    // ---- online softmax per slab (exp2 domain): max via permlane16,
    //      row sums via WMMA against the all-ones matrix ----
    h16 pa[2];
#pragma unroll
    for (int s = 0; s < 2; ++s) {
      _Float16* sPw = &sP[wid][s][0];
#pragma unroll
      for (int r = 0; r < 8; ++r) {
        float rm   = red_max16(fmaxf(c0[s][r], c1[s][r]));
        float mn   = fmaxf(mrow[s][r], rm);
        float corr = __builtin_amdgcn_exp2f(mrow[s][r] - mn);
        float p0   = __builtin_amdgcn_exp2f(c0[s][r] - mn);
        float p1   = __builtin_amdgcn_exp2f(c1[s][r] - mn);
        mrow[s][r] = mn;
        lrow[s][r] *= corr;
        const int row = r + 8 * hi;
        sPw[row * 32 + ml]      = (_Float16)p0;
        sPw[row * 32 + 16 + ml] = (_Float16)p1;
#pragma unroll
        for (int dt = 0; dt < 8; ++dt) o[s][dt][r] *= corr;
      }
      // P back into A layout
      const _Float16* pb = &sPw[ml * 32 + hi * 8];
      h8 plo = *(const h8*)(pb);
      h8 phi = *(const h8*)(pb + 16);
#pragma unroll
      for (int i = 0; i < 8; ++i) { pa[s][i] = plo[i]; pa[s][8 + i] = phi[i]; }

      // row sums on the matrix pipe: C(M,N) = sum_K P(M,K) * 1 = rowsum(M)
      f8 rs = {};
      rs = __builtin_amdgcn_wmma_f32_16x16x32_f16(false, pa[s], false, ones,
                                                  (short)0, rs, false, false);
#pragma unroll
      for (int r = 0; r < 8; ++r) lrow[s][r] += rs[r];
    }

    // ---- O += P . V : 8 shared V tiles feed both slabs (16 WMMAs) ----
    h16 vt[8];
#pragma unroll
    for (int dt = 0; dt < 8; ++dt) {
      h8 vl = *(const h8*)(&sVt[(dt * 16 + ml) * KT + hi * 8]);
      h8 vh = *(const h8*)(&sVt[(dt * 16 + ml) * KT + 16 + hi * 8]);
#pragma unroll
      for (int i = 0; i < 8; ++i) { vt[dt][i] = vl[i]; vt[dt][8 + i] = vh[i]; }
    }
#pragma unroll
    for (int dt = 0; dt < 8; ++dt) {
#pragma unroll
      for (int s = 0; s < 2; ++s)
        o[s][dt] = __builtin_amdgcn_wmma_f32_16x16x32_f16(false, pa[s], false, vt[dt],
                                                          (short)0, o[s][dt], false, false);
    }
    __syncthreads();
  }

  // ---- epilogue: normalize and scatter to (B, L, H*D) ----
#pragma unroll
  for (int s = 0; s < 2; ++s) {
#pragma unroll
    for (int r = 0; r < 8; ++r) {
      const int row = q0 + s * 16 + r + 8 * hi;
      float inv = 1.0f / lrow[s][r];
      float* op = out + ((size_t)b * L_ + row) * (H_ * D_) + h * D_;
#pragma unroll
      for (int dt = 0; dt < 8; ++dt)
        op[dt * 16 + ml] = o[s][dt][r] * inv;
    }
  }
}

extern "C" void kernel_launch(void* const* d_in, const int* in_sizes, int n_in,
                              void* d_out, int out_size, void* d_ws, size_t ws_size,
                              hipStream_t stream) {
  const float* q  = (const float*)d_in[0];
  const float* k  = (const float*)d_in[1];
  const float* v  = (const float*)d_in[2];
  const float* pe = (const float*)d_in[3];
  float* out = (float*)d_out;

  dim3 grid(B_ * H_ * NQB);   // 2*24*16 = 768 workgroups
  dim3 block(128);            // 4 waves (wave32)
  hipLaunchKernelGGL(flux_rope_attn_kernel, grid, block, 0, stream,
                     q, k, v, pe, out);
}
